// MultiHeadSelfAttention_53111565582457
// MI455X (gfx1250) — matrix operand order repaired and checked by reading, not compiled, not run on hardware
//
#include <hip/hip_runtime.h>
#include <hip/hip_bf16.h>

typedef __attribute__((ext_vector_type(16))) _Float16 v16h;
typedef __attribute__((ext_vector_type(8)))  float    v8f;

#define D_MODEL 768
#define N_HEADS 12
#define D_HEAD  64
#define BATCH   2
#define SEQ     2048
#define BT      (BATCH * SEQ)     // 4096
#define NQKV    (3 * D_MODEL)     // 2304

// ---------------------------------------------------------------------------
// CDNA5 WMMA fragment gathers (wave32). Layouts per CDNA5 ISA:
//  A (f16, 16x32): lanes 0-15 -> M=lane, lanes 16-31 -> M=lane-16.
//    VGPR i (i<4):  K = 8*half + 2i, 2i+1 ; VGPR i (i>=4): K = 16 + 8*half + 2(i-4)
//  B (f16, 32x16): lanes hold N=lane&15; same K order as A
//  C/D (f32, 16x16): VGPR r: M = r + 8*half, N = lane&15
// ---------------------------------------------------------------------------
__device__ __forceinline__ v16h load_a_frag(const _Float16* base, int ld) {
  const int lane = threadIdx.x & 31;
  const int m  = lane & 15;
  const int hh = lane >> 4;
  const _Float16* row = base + m * ld;
  v16h a;
#pragma unroll
  for (int i = 0; i < 4; ++i) {
    const int k = 8 * hh + 2 * i;
    a[2 * i + 0] = row[k + 0];
    a[2 * i + 1] = row[k + 1];
  }
#pragma unroll
  for (int i = 0; i < 4; ++i) {
    const int k = 16 + 8 * hh + 2 * i;
    a[8 + 2 * i + 0] = row[k + 0];
    a[8 + 2 * i + 1] = row[k + 1];
  }
  return a;
}

// Storage convention: row = N (16 rows), contiguous K (32) -> ds_load_b128 x4.
__device__ __forceinline__ v16h load_b_frag(const _Float16* base, int ld) {
  const int lane = threadIdx.x & 31;
  const int n  = lane & 15;
  const int hh = lane >> 4;
  const _Float16* row = base + n * ld;
  v16h b;
#pragma unroll
  for (int v = 0; v < 8; ++v) {
    const int k = (v < 4 ? 0 : 8) + 8 * hh + 2 * v;
    b[2 * v + 0] = row[k + 0];
    b[2 * v + 1] = row[k + 1];
  }
  return b;
}

__device__ __forceinline__ v8f wmma_f32_f16(v16h a, v16h b, v8f c) {
  // 8 args: (neg_a, A, neg_b, B, c_mod, C, reuse_a, reuse_b)
  return __builtin_amdgcn_wmma_f32_16x16x32_f16(false, a, false, b, (short)0, c,
                                                false, false);
}

// gfx1250 async copy: global -> LDS, 16 bytes/lane, tracked by ASYNCcnt.
// VDST carries the LDS byte offset (low 32 bits of the generic shared pointer).
__device__ __forceinline__ void async_load_b128(void* lds, const void* gptr) {
  const unsigned lds_off = (unsigned)(size_t)lds;
  asm volatile("global_load_async_to_lds_b128 %0, %1, off"
               :: "v"(lds_off), "v"(gptr)
               : "memory");
}
__device__ __forceinline__ void wait_async0() {
  asm volatile("s_wait_asynccnt 0" ::: "memory");
}

// ---------------------------------------------------------------------------
// Kernel 1: qkv = x @ Wqkv^T + b  (M=4096, N=2304, K=768), scatter to
//   Q,K : [B*H][T][64] f16       V : [B*H][64][T] f16 (transposed)
// block = 256 thr (8 waves), 128x64 block tile, BK=32.
// Waves: 4 row-strips x 2 col-strips; each wave 32x32 -> 4 WMMA / K-step.
// ---------------------------------------------------------------------------
__global__ __launch_bounds__(256) void qkv_gemm_kernel(
    const float* __restrict__ x, const float* __restrict__ Wqkv,
    const float* __restrict__ bqkv, _Float16* __restrict__ Qh,
    _Float16* __restrict__ Kh, _Float16* __restrict__ Vt) {
  __shared__ _Float16 As[128][40];
  __shared__ _Float16 Bs[64][40];
  const int tid  = threadIdx.x;
  const int lane = tid & 31;
  const int wid  = tid >> 5;
  const int wy   = wid & 3;   // 32-row strip
  const int wx   = wid >> 2;  // 32-col strip
  const int gm0  = blockIdx.x * 128;
  const int gn0  = blockIdx.y * 64;

  v8f acc[2][2] = {};
  for (int k0 = 0; k0 < D_MODEL; k0 += 32) {
#pragma unroll
    for (int i = 0; i < 16; ++i) {  // A: 128x32 f32 -> f16
      const int idx = tid + 256 * i;
      const int r = idx >> 5, c = idx & 31;
      As[r][c] = (_Float16)x[(gm0 + r) * D_MODEL + k0 + c];
    }
#pragma unroll
    for (int i = 0; i < 8; ++i) {   // B[k][n] = W[n][k]: 64x32 f32 -> f16
      const int idx = tid + 256 * i;
      const int r = idx >> 5, c = idx & 31;
      Bs[r][c] = (_Float16)Wqkv[(gn0 + r) * D_MODEL + k0 + c];
    }
    __syncthreads();
    const v16h a0 = load_a_frag(&As[32 * wy + 0][0], 40);
    const v16h a1 = load_a_frag(&As[32 * wy + 16][0], 40);
    const v16h b0 = load_b_frag(&Bs[32 * wx + 0][0], 40);
    const v16h b1 = load_b_frag(&Bs[32 * wx + 16][0], 40);
    acc[0][0] = wmma_f32_f16(a0, b0, acc[0][0]);
    acc[0][1] = wmma_f32_f16(a0, b1, acc[0][1]);
    acc[1][0] = wmma_f32_f16(a1, b0, acc[1][0]);
    acc[1][1] = wmma_f32_f16(a1, b1, acc[1][1]);
    __syncthreads();
  }

  const int hh = lane >> 4;
  const int nl = lane & 15;
#pragma unroll
  for (int mi = 0; mi < 2; ++mi) {
#pragma unroll
    for (int jt = 0; jt < 2; ++jt) {
      const int n = gn0 + 32 * wx + 16 * jt + nl;
      const float bias = bqkv[n];
#pragma unroll
      for (int r = 0; r < 8; ++r) {
        const int m = gm0 + 32 * wy + 16 * mi + r + 8 * hh;
        const float val = acc[mi][jt][r] + bias;
        const int bb = m >> 11;        // / SEQ
        const int t  = m & (SEQ - 1);  // % SEQ
        if (n < D_MODEL) {
          const int h = n >> 6, d = n & 63;
          Qh[((size_t)(bb * N_HEADS + h) * SEQ + t) * D_HEAD + d] = (_Float16)val;
        } else if (n < 2 * D_MODEL) {
          const int nn = n - D_MODEL;
          const int h = nn >> 6, d = nn & 63;
          Kh[((size_t)(bb * N_HEADS + h) * SEQ + t) * D_HEAD + d] = (_Float16)val;
        } else {
          const int nn = n - 2 * D_MODEL;
          const int h = nn >> 6, d = nn & 63;
          Vt[((size_t)(bb * N_HEADS + h) * D_HEAD + d) * SEQ + t] = (_Float16)val;
        }
      }
    }
  }
}

// ---------------------------------------------------------------------------
// Kernel 2: causal flash attention. grid=(T/64, B*H), block=128 (4 waves).
// Each wave owns a 16-query strip; block iterates 64-key tiles staged in LDS
// via gfx1250 async-to-LDS copies (ASYNCcnt). 16 WMMA per tile per wave.
// ---------------------------------------------------------------------------
__global__ __launch_bounds__(128) void attn_kernel(
    const _Float16* __restrict__ Qh, const _Float16* __restrict__ Kh,
    const _Float16* __restrict__ Vt, _Float16* __restrict__ Ao) {
  __shared__ _Float16 Qs[64][72];      // queries x d
  __shared__ _Float16 Ks[64][72];      // keys x d
  __shared__ _Float16 Vs[64][72];      // d x keys (V pre-transposed in HBM)
  __shared__ _Float16 Ps[4][16][72];   // per-wave P transpose bounce
  const int tid  = threadIdx.x;
  const int lane = tid & 31;
  const int wid  = tid >> 5;
  const int q0   = blockIdx.x * 64;
  const int bh   = blockIdx.y;
  const int bb   = bh / N_HEADS;
  const int h    = bh % N_HEADS;
  const _Float16* Qg = Qh + (size_t)bh * SEQ * D_HEAD;
  const _Float16* Kg = Kh + (size_t)bh * SEQ * D_HEAD;
  const _Float16* Vg = Vt + (size_t)bh * D_HEAD * SEQ;

  // async-stage 64x64 Q tile (8 halves = 16B per op, all 16B-aligned)
#pragma unroll
  for (int i = 0; i < 4; ++i) {
    const int idx = tid * 8 + 1024 * i;
    const int r = idx >> 6, c = idx & 63;
    async_load_b128(&Qs[r][c], Qg + (size_t)(q0 + r) * D_HEAD + c);
  }
  wait_async0();
  __syncthreads();
  const int qs = 16 * wid;
  const v16h aq0 = load_a_frag(&Qs[qs][0], 72);
  const v16h aq1 = load_a_frag(&Qs[qs][32], 72);

  v8f o[4] = {};
  float mrow[8], lrow[8];
#pragma unroll
  for (int r = 0; r < 8; ++r) { mrow[r] = -1e30f; lrow[r] = 0.0f; }

  const int hh = lane >> 4;
  const int nl = lane & 15;
  const int nkt = (q0 + 64) / 64;  // causal extent in 64-key tiles
  for (int kt = 0; kt < nkt; ++kt) {
    const int k0 = kt * 64;
    // async-stage K (64 keys x 64 d) and V^T (64 d x 64 keys)
#pragma unroll
    for (int i = 0; i < 4; ++i) {
      const int idx = tid * 8 + 1024 * i;
      const int r = idx >> 6, c = idx & 63;
      async_load_b128(&Ks[r][c], Kg + (size_t)(k0 + r) * D_HEAD + c);
      async_load_b128(&Vs[r][c], Vg + (size_t)r * SEQ + k0 + c);
    }
    wait_async0();
    __syncthreads();

    // S = (Q K^T)/sqrt(d): four 16-key n-tiles x two 32-d k-steps = 8 WMMA
    v8f s[4] = {};
#pragma unroll
    for (int nt = 0; nt < 4; ++nt) {
      const v16h bk0 = load_b_frag(&Ks[16 * nt][0], 72);
      const v16h bk1 = load_b_frag(&Ks[16 * nt][32], 72);
      s[nt] = wmma_f32_f16(aq0, bk0, s[nt]);
      s[nt] = wmma_f32_f16(aq1, bk1, s[nt]);
    }
    // scale + causal mask
#pragma unroll
    for (int nt = 0; nt < 4; ++nt) {
#pragma unroll
      for (int r = 0; r < 8; ++r) {
        const int qrow = q0 + qs + r + 8 * hh;
        const int key  = k0 + 16 * nt + nl;
        const float v = s[nt][r] * 0.125f;  // 1/sqrt(64)
        s[nt][r] = (key > qrow) ? -1e30f : v;
      }
    }
    // row max over 64 keys: combine n-tiles, reduce over the 16-lane half
    v8f rmx;
#pragma unroll
    for (int r = 0; r < 8; ++r)
      rmx[r] = fmaxf(fmaxf(s[0][r], s[1][r]), fmaxf(s[2][r], s[3][r]));
#pragma unroll
    for (int off = 1; off <= 8; off <<= 1)
#pragma unroll
      for (int r = 0; r < 8; ++r)
        rmx[r] = fmaxf(rmx[r], __shfl_xor(rmx[r], off, 32));

    float alpha[8];
    v8f rsum;
#pragma unroll
    for (int r = 0; r < 8; ++r) {
      const float mnew = fmaxf(mrow[r], rmx[r]);
      alpha[r] = __expf(mrow[r] - mnew);
      mrow[r] = mnew;
      float acc = 0.0f;
#pragma unroll
      for (int nt = 0; nt < 4; ++nt) {
        const float p = __expf(s[nt][r] - mnew);
        s[nt][r] = p;
        acc += p;
      }
      rsum[r] = acc;
    }
#pragma unroll
    for (int off = 1; off <= 8; off <<= 1)
#pragma unroll
      for (int r = 0; r < 8; ++r)
        rsum[r] += __shfl_xor(rsum[r], off, 32);
#pragma unroll
    for (int r = 0; r < 8; ++r) lrow[r] = lrow[r] * alpha[r] + rsum[r];
#pragma unroll
    for (int j = 0; j < 4; ++j)
#pragma unroll
      for (int r = 0; r < 8; ++r) o[j][r] *= alpha[r];

    // transpose P (C-layout -> 16x64 row-major) via per-wave LDS bounce
#pragma unroll
    for (int nt = 0; nt < 4; ++nt)
#pragma unroll
      for (int r = 0; r < 8; ++r)
        Ps[wid][r + 8 * hh][16 * nt + nl] = (_Float16)s[nt][r];
    __syncthreads();
    const v16h ap0 = load_a_frag(&Ps[wid][0][0], 72);
    const v16h ap1 = load_a_frag(&Ps[wid][0][32], 72);
    // O += P @ V: four d n-tiles x two 32-key k-steps = 8 WMMA
#pragma unroll
    for (int j = 0; j < 4; ++j) {
      const v16h bv0 = load_b_frag(&Vs[16 * j][0], 72);
      const v16h bv1 = load_b_frag(&Vs[16 * j][32], 72);
      o[j] = wmma_f32_f16(ap0, bv0, o[j]);
      o[j] = wmma_f32_f16(ap1, bv1, o[j]);
    }
    __syncthreads();
  }

  // normalize and store to Ao [B*T][768] f16 (proj-GEMM friendly)
#pragma unroll
  for (int r = 0; r < 8; ++r) {
    const float inv = 1.0f / lrow[r];
    const int t = q0 + qs + r + 8 * hh;
    const size_t row = (size_t)bb * SEQ + t;
#pragma unroll
    for (int j = 0; j < 4; ++j) {
      const int col = h * D_HEAD + 16 * j + nl;
      Ao[row * D_MODEL + col] = (_Float16)(o[j][r] * inv);
    }
  }
}

// ---------------------------------------------------------------------------
// Kernel 3: out = Ao @ Wproj^T + b  (M=4096, N=768, K=768), f32 output.
// Same tiling as kernel 1; f16 A-operand staged with async-to-LDS copies.
// ---------------------------------------------------------------------------
__global__ __launch_bounds__(256) void proj_gemm_kernel(
    const _Float16* __restrict__ Ain, const float* __restrict__ Wp,
    const float* __restrict__ bp, float* __restrict__ out) {
  __shared__ _Float16 As[128][40];
  __shared__ _Float16 Bs[64][40];
  const int tid  = threadIdx.x;
  const int lane = tid & 31;
  const int wid  = tid >> 5;
  const int wy   = wid & 3;
  const int wx   = wid >> 2;
  const int gm0  = blockIdx.x * 128;
  const int gn0  = blockIdx.y * 64;

  v8f acc[2][2] = {};
  for (int k0 = 0; k0 < D_MODEL; k0 += 32) {
#pragma unroll
    for (int i = 0; i < 2; ++i) {   // A: 128x32 f16, async 16B/lane
      const int idx = tid * 8 + 2048 * i;
      const int r = idx >> 5, c = idx & 31;
      async_load_b128(&As[r][c], Ain + (size_t)(gm0 + r) * D_MODEL + k0 + c);
    }
#pragma unroll
    for (int i = 0; i < 8; ++i) {   // B[k][n] = Wp[n][k]: f32 -> f16
      const int idx = tid + 256 * i;
      const int r = idx >> 5, c = idx & 31;
      Bs[r][c] = (_Float16)Wp[(gn0 + r) * D_MODEL + k0 + c];
    }
    wait_async0();
    __syncthreads();
    const v16h a0 = load_a_frag(&As[32 * wy + 0][0], 40);
    const v16h a1 = load_a_frag(&As[32 * wy + 16][0], 40);
    const v16h b0 = load_b_frag(&Bs[32 * wx + 0][0], 40);
    const v16h b1 = load_b_frag(&Bs[32 * wx + 16][0], 40);
    acc[0][0] = wmma_f32_f16(a0, b0, acc[0][0]);
    acc[0][1] = wmma_f32_f16(a0, b1, acc[0][1]);
    acc[1][0] = wmma_f32_f16(a1, b0, acc[1][0]);
    acc[1][1] = wmma_f32_f16(a1, b1, acc[1][1]);
    __syncthreads();
  }

  const int hh = lane >> 4;
  const int nl = lane & 15;
#pragma unroll
  for (int mi = 0; mi < 2; ++mi) {
#pragma unroll
    for (int jt = 0; jt < 2; ++jt) {
      const int n = gn0 + 32 * wx + 16 * jt + nl;
      const float bias = bp[n];
#pragma unroll
      for (int r = 0; r < 8; ++r) {
        const int m = gm0 + 32 * wy + 16 * mi + r + 8 * hh;
        out[(size_t)m * D_MODEL + n] = acc[mi][jt][r] + bias;
      }
    }
  }
}

// ---------------------------------------------------------------------------
extern "C" void kernel_launch(void* const* d_in, const int* in_sizes, int n_in,
                              void* d_out, int out_size, void* d_ws, size_t ws_size,
                              hipStream_t stream) {
  const float* x     = (const float*)d_in[0];
  // d_in[1] = causal mask (int32) -- applied analytically, not read
  const float* Wqkv  = (const float*)d_in[2];
  const float* bqkv  = (const float*)d_in[3];
  const float* Wproj = (const float*)d_in[4];
  const float* bproj = (const float*)d_in[5];
  float* out = (float*)d_out;

  _Float16* ws = (_Float16*)d_ws;
  const size_t perQKV = (size_t)BATCH * N_HEADS * SEQ * D_HEAD;  // 3,145,728 halves
  _Float16* Qh = ws;                  // [B*H][T][64]
  _Float16* Kh = Qh + perQKV;         // [B*H][T][64]
  _Float16* Vt = Kh + perQKV;         // [B*H][64][T]
  _Float16* Ao = Vt + perQKV;         // [B*T][768]

  dim3 g1(BT / 128, NQKV / 64);       // (32, 36)
  qkv_gemm_kernel<<<g1, 256, 0, stream>>>(x, Wqkv, bqkv, Qh, Kh, Vt);

  dim3 g2(SEQ / 64, BATCH * N_HEADS); // (32, 24)
  attn_kernel<<<g2, 128, 0, stream>>>(Qh, Kh, Vt, Ao);

  dim3 g3(BT / 128, D_MODEL / 64);    // (32, 12)
  proj_gemm_kernel<<<g3, 256, 0, stream>>>(Ao, Wproj, bproj, out);
}
